// GATConvNet_22514218566314
// MI455X (gfx1250) — matrix operand order repaired and checked by reading, not compiled, not run on hardware
//
#include <hip/hip_runtime.h>
#include <hip/hip_bf16.h>

typedef __attribute__((ext_vector_type(16))) _Float16 v16h;
typedef __attribute__((ext_vector_type(8)))  float    v8f;

#define HEADS 4
#define CPH   64
#define HIDD  256   // HEADS*CPH
#define TK    32    // K-step per WMMA
#define NEG_SLOPE 0.2f

// ---------------- utility kernels ----------------

__global__ void k_fill_f32(float* __restrict__ p, float v, size_t n) {
    size_t i = (size_t)blockIdx.x * blockDim.x + threadIdx.x;
    size_t stride = (size_t)gridDim.x * blockDim.x;
    for (; i < n; i += stride) p[i] = v;
}

__global__ void k_fill_u32(unsigned* __restrict__ p, unsigned v, size_t n) {
    size_t i = (size_t)blockIdx.x * blockDim.x + threadIdx.x;
    size_t stride = (size_t)gridDim.x * blockDim.x;
    for (; i < n; i += stride) p[i] = v;
}

__global__ void k_cvt_f32_f16(const float* __restrict__ in, _Float16* __restrict__ out, size_t n) {
    size_t i = (size_t)blockIdx.x * blockDim.x + threadIdx.x;
    size_t stride = (size_t)gridDim.x * blockDim.x;
    for (; i < n; i += stride) out[i] = (_Float16)in[i];
}

// ---------------- WMMA GEMM with async-to-LDS staging ----------------
// Block = 128 threads (4 waves). All waves share one 16-row M tile; wave w owns
// the 64-column group w. Per 32-K step, A chunk (16x32, 1KB) and B chunk
// (32x256, 16KB) are staged into LDS with GLOBAL_LOAD_ASYNC_TO_LDS_B128
// (ASYNCcnt, drained by s_wait_asynccnt before the barrier), then each wave
// issues 4 back-to-back v_wmma_f32_16x16x32_f16 off one LDS A fragment.
__global__ void k_wmma_gemm(const _Float16* __restrict__ A,
                            const _Float16* __restrict__ B,
                            const float*    __restrict__ bias,
                            float*          __restrict__ C,
                            int M, int K, int Nout /* == HIDD */) {
    __shared__ _Float16 As[16 * TK];      //  1 KB
    __shared__ _Float16 Bs[TK * HIDD];    // 16 KB

    const int tid  = threadIdx.x;         // 0..127
    const int lane = tid & 31;
    const int wv   = tid >> 5;            // 0..3: 64-col group
    const int tm   = blockIdx.x << 4;
    const int tn0  = wv << 6;
    if (tm >= M) return;                  // block-uniform

    const int half = lane >> 4;           // 0: lanes 0-15, 1: lanes 16-31
    const int l16  = lane & 15;

    v8f acc0 = {}, acc1 = {}, acc2 = {}, acc3 = {};

    for (int k0 = 0; k0 < K; k0 += TK) {
        // --- stage A chunk: 64 async b128 (threads 0..63, one each) ---
        if (tid < 64) {
            int r  = tid >> 2;            // row 0..15
            int cq = (tid & 3) << 3;      // 8-half col offset
            unsigned lds_off = (unsigned)(uintptr_t)&As[r * TK + cq];
            const _Float16* g = A + (size_t)(tm + r) * K + k0 + cq;
            asm volatile("global_load_async_to_lds_b128 %0, %1, off"
                         :: "v"(lds_off), "v"((unsigned long long)(uintptr_t)g)
                         : "memory");
        }
        // --- stage B chunk: 1024 async b128 (128 threads x 8) ---
        #pragma unroll
        for (int it = 0; it < 8; ++it) {
            int idx = tid + it * 128;     // 0..1023
            int r   = idx >> 5;           // row 0..31
            int cq  = (idx & 31) << 3;    // col offset 0..248
            unsigned lds_off = (unsigned)(uintptr_t)&Bs[r * HIDD + cq];
            const _Float16* g = B + (size_t)(k0 + r) * Nout + cq;
            asm volatile("global_load_async_to_lds_b128 %0, %1, off"
                         :: "v"(lds_off), "v"((unsigned long long)(uintptr_t)g)
                         : "memory");
        }
        asm volatile("s_wait_asynccnt 0x0" ::: "memory");
        __syncthreads();

        // --- fragments from LDS ---
        v16h a = *(const v16h*)&As[l16 * TK + half * 16];
        const _Float16* brow = &Bs[(half * 16) * HIDD + tn0 + l16];
        v16h b0, b1, b2, b3;
        #pragma unroll
        for (int s = 0; s < 16; ++s) {
            b0[s] = brow[s * HIDD];
            b1[s] = brow[s * HIDD + 16];
            b2[s] = brow[s * HIDD + 32];
            b3[s] = brow[s * HIDD + 48];
        }
        acc0 = __builtin_amdgcn_wmma_f32_16x16x32_f16(false, a, false, b0, (short)0, acc0, false, false);
        acc1 = __builtin_amdgcn_wmma_f32_16x16x32_f16(false, a, false, b1, (short)0, acc1, false, false);
        acc2 = __builtin_amdgcn_wmma_f32_16x16x32_f16(false, a, false, b2, (short)0, acc2, false, false);
        acc3 = __builtin_amdgcn_wmma_f32_16x16x32_f16(false, a, false, b3, (short)0, acc3, false, false);
        __syncthreads();
    }

    // C/D layout: VGPR v, lane -> (M = v + 8*half, N = l16)
    #pragma unroll
    for (int v = 0; v < 8; ++v) {
        size_t row = (size_t)(tm + v + 8 * half) * Nout;
        int col = tn0 + l16;
        C[row + col +  0] = acc0[v] + bias[col +  0];
        C[row + col + 16] = acc1[v] + bias[col + 16];
        C[row + col + 32] = acc2[v] + bias[col + 32];
        C[row + col + 48] = acc3[v] + bias[col + 48];
    }
}

// ---------------- edge passes ----------------

__device__ __forceinline__ unsigned enc_f32(float x) {
    unsigned u = __float_as_uint(x);
    return (u & 0x80000000u) ? ~u : (u | 0x80000000u);
}
__device__ __forceinline__ float dec_f32(unsigned k) {
    unsigned u = (k & 0x80000000u) ? (k ^ 0x80000000u) : ~k;
    return __uint_as_float(u);
}

// Pass A: per (edge, head) leaky-relu'd attention logit + segment max via encoded atomicMax
__global__ void k_edge_logits(const float* __restrict__ xl, const float* __restrict__ xr,
                              const int* __restrict__ ei, const float* __restrict__ att,
                              float* __restrict__ e, unsigned* __restrict__ nmax,
                              int E, int Nn) {
    long long t = (long long)blockIdx.x * blockDim.x + threadIdx.x;
    long long total = (long long)(E + Nn) * HEADS;
    if (t >= total) return;
    int eid = (int)(t >> 2), h = (int)(t & 3);
    int src, dst;
    if (eid < E) { src = ei[eid]; dst = ei[(size_t)E + eid]; }
    else         { src = dst = eid - E; }
    const float* pl = xl + (size_t)src * HIDD + h * CPH;
    const float* pr = xr + (size_t)dst * HIDD + h * CPH;
    const float* pa = att + h * CPH;
    float s = 0.f;
    #pragma unroll 8
    for (int c = 0; c < CPH; ++c) {
        float m = pl[c] + pr[c];
        m = (m > 0.f) ? m : NEG_SLOPE * m;
        s += m * pa[c];
    }
    e[t] = s;
    atomicMax(&nmax[(size_t)dst * HEADS + h], enc_f32(s));
}

// Pass B: exponentiate against segment max, accumulate denominators
__global__ void k_edge_exp(const int* __restrict__ ei, float* __restrict__ e,
                           const unsigned* __restrict__ nmax, float* __restrict__ denom,
                           int E, int Nn) {
    long long t = (long long)blockIdx.x * blockDim.x + threadIdx.x;
    long long total = (long long)(E + Nn) * HEADS;
    if (t >= total) return;
    int eid = (int)(t >> 2), h = (int)(t & 3);
    int dst = (eid < E) ? ei[(size_t)E + eid] : (eid - E);
    float mx = dec_f32(nmax[(size_t)dst * HEADS + h]);
    float ex = __expf(e[t] - mx);
    e[t] = ex;
    atomicAdd(&denom[(size_t)dst * HEADS + h], ex);
}

// Pass C: alpha-weighted aggregation. One 256-thread block per edge; thread = channel.
__global__ void k_edge_aggregate(const float* __restrict__ xl, const int* __restrict__ ei,
                                 const float* __restrict__ e, const float* __restrict__ denom,
                                 float* __restrict__ accum, int E, int Nn) {
    int c = threadIdx.x;          // 0..255
    int h = c >> 6;
    for (int eid = blockIdx.x; eid < E + Nn; eid += gridDim.x) {
        int src, dst;
        if (eid < E) { src = ei[eid]; dst = ei[(size_t)E + eid]; }
        else         { src = dst = eid - E; }
        float alpha = e[(size_t)eid * HEADS + h] /
                      (denom[(size_t)dst * HEADS + h] + 1e-16f);
        atomicAdd(&accum[(size_t)dst * HIDD + c], alpha * xl[(size_t)src * HIDD + c]);
    }
}

// Finalize (concat layers): h = ELU(accum + bias), in place
__global__ void k_finalize_concat(float* __restrict__ accum, const float* __restrict__ bias, int Nn) {
    size_t t = (size_t)blockIdx.x * blockDim.x + threadIdx.x;
    size_t total = (size_t)Nn * HIDD;
    if (t >= total) return;
    float v = accum[t] + bias[t & (HIDD - 1)];
    accum[t] = (v > 0.f) ? v : (__expf(v) - 1.f);
}

// Finalize (last layer): mean over heads + bias + ELU -> d_out [N, 64]
__global__ void k_finalize_mean(const float* __restrict__ accum, const float* __restrict__ bias,
                                float* __restrict__ out, int Nn) {
    size_t t = (size_t)blockIdx.x * blockDim.x + threadIdx.x;
    size_t total = (size_t)Nn * CPH;
    if (t >= total) return;
    size_t i = t >> 6;
    int c = (int)(t & (CPH - 1));
    const float* a = accum + i * HIDD;
    float s = 0.25f * (a[c] + a[CPH + c] + a[2 * CPH + c] + a[3 * CPH + c]) + bias[c];
    out[t] = (s > 0.f) ? s : (__expf(s) - 1.f);
}

// ---------------- host driver ----------------

static inline int cdiv(long long a, int b) { return (int)((a + b - 1) / b); }

extern "C" void kernel_launch(void* const* d_in, const int* in_sizes, int n_in,
                              void* d_out, int out_size, void* d_ws, size_t ws_size,
                              hipStream_t stream) {
    const int IN = 128;
    const int Nn = in_sizes[0] / IN;        // 20000
    const int E  = in_sizes[1] / 2;         // 640000
    const int Etot = E + Nn;

    const float* x  = (const float*)d_in[0];
    const int*   ei = (const int*)d_in[1];

    // workspace layout (all offsets multiples of 64B)
    char* ws = (char*)d_ws;
    size_t nh = (size_t)Nn * HIDD;
    float*     xl    = (float*)ws;                      ws += nh * sizeof(float);
    float*     xr    = (float*)ws;                      ws += nh * sizeof(float);
    float*     hbuf  = (float*)ws;                      ws += nh * sizeof(float);   // accumulator / activations
    float*     e     = (float*)ws;                      ws += (size_t)Etot * HEADS * sizeof(float);
    float*     denom = (float*)ws;                      ws += (size_t)Nn * HEADS * sizeof(float);
    unsigned*  nmax  = (unsigned*)ws;                   ws += (size_t)Nn * HEADS * sizeof(unsigned);
    _Float16*  h16   = (_Float16*)ws;                   ws += nh * sizeof(_Float16);
    _Float16*  wl16  = (_Float16*)ws;                   ws += (size_t)HIDD * HIDD * sizeof(_Float16);
    _Float16*  wr16  = (_Float16*)ws;

    const unsigned NEG_INF_KEY = 0x007FFFFFu;           // enc(-inf)

    for (int l = 0; l < 3; ++l) {
        const int Kdim = (l == 0) ? IN : HIDD;
        const float* hin  = (l == 0) ? x : hbuf;
        const int base = 2 + 6 * l;
        const float* Wl   = (const float*)d_in[base + 0];
        const float* Wr   = (const float*)d_in[base + 1];
        const float* bl   = (const float*)d_in[base + 2];
        const float* br   = (const float*)d_in[base + 3];
        const float* att  = (const float*)d_in[base + 4];
        const float* bias = (const float*)d_in[base + 5];

        // f32 -> f16 staging
        size_t na = (size_t)Nn * Kdim;
        size_t nw = (size_t)Kdim * HIDD;
        k_cvt_f32_f16<<<2048, 256, 0, stream>>>(hin, h16, na);
        k_cvt_f32_f16<<<256, 256, 0, stream>>>(Wl, wl16, nw);
        k_cvt_f32_f16<<<256, 256, 0, stream>>>(Wr, wr16, nw);

        // GEMMs: one 16-row M tile per 128-thread block (4 waves x 64-col groups)
        dim3 gb(Nn / 16), bb(128);
        k_wmma_gemm<<<gb, bb, 0, stream>>>(h16, wl16, bl, xl, Nn, Kdim, HIDD);
        k_wmma_gemm<<<gb, bb, 0, stream>>>(h16, wr16, br, xr, Nn, Kdim, HIDD);

        // init segment-softmax state + accumulator
        k_fill_f32<<<2048, 256, 0, stream>>>(hbuf, 0.f, nh);
        k_fill_f32<<<256, 256, 0, stream>>>(denom, 0.f, (size_t)Nn * HEADS);
        k_fill_u32<<<256, 256, 0, stream>>>(nmax, NEG_INF_KEY, (size_t)Nn * HEADS);

        // edge passes
        long long eh = (long long)Etot * HEADS;
        k_edge_logits<<<cdiv(eh, 256), 256, 0, stream>>>(xl, xr, ei, att, e, nmax, E, Nn);
        k_edge_exp<<<cdiv(eh, 256), 256, 0, stream>>>(ei, e, nmax, denom, E, Nn);
        k_edge_aggregate<<<Etot, 256, 0, stream>>>(xl, ei, e, denom, hbuf, E, Nn);

        if (l < 2) {
            k_finalize_concat<<<cdiv((long long)Nn * HIDD, 256), 256, 0, stream>>>(hbuf, bias, Nn);
        } else {
            k_finalize_mean<<<cdiv((long long)Nn * CPH, 256), 256, 0, stream>>>(hbuf, bias, (float*)d_out, Nn);
        }
    }
}
